// LSTMDecoderWithAttention_20804821582004
// MI455X (gfx1250) — compile-verified, matrix operand order checked by reading, not run on hardware
//
#include <hip/hip_runtime.h>
#include <cstdint>

#define DEV __device__ __forceinline__

typedef __attribute__((ext_vector_type(16))) __bf16 bf16x16;
typedef __attribute__((ext_vector_type(8)))  float  f32x8;

// ---------------------------------------------------------------- helpers ---
DEV uint16_t bf16_bits(float f) {            // round-to-nearest-even fp32->bf16
  uint32_t u = __float_as_uint(f);
  u += 0x7fffu + ((u >> 16) & 1u);
  return (uint16_t)(u >> 16);
}

DEV float sigmoidf_(float x) { return 1.0f / (1.0f + __expf(-x)); }

// Load a 16x32 bf16 fragment (A layout per CDNA5 ISA 7.12.2) from a row-major
// matrix: row = lane&15, K-halves selected by lane>=16.  Same loader serves
// the B fragment of C = A * W^T, since B[k][n] = W[n][k] (row n, stride K).
DEV bf16x16 load_frag(const uint16_t* tile, int ld) {
  const int lane = (int)(threadIdx.x & 31u);
  const int r    = lane & 15;
  const int hi   = lane >> 4;
  const uint32_t* p = reinterpret_cast<const uint32_t*>(tile + (size_t)r * ld);
  union { bf16x16 v; uint32_t d[8]; } u;
#pragma unroll
  for (int i = 0; i < 4; ++i) u.d[i]     = p[(hi << 2) + i];       // K 0..7 / 8..15
#pragma unroll
  for (int i = 0; i < 4; ++i) u.d[4 + i] = p[8 + (hi << 2) + i];   // K 16..23 / 24..31
  return u.v;
}

// ------------------------------------------------------- conversion / init ---
__global__ void cvt_bf16_kernel(const float* __restrict__ src,
                                uint16_t* __restrict__ dst, long n) {
  long i = (long)blockIdx.x * blockDim.x + threadIdx.x;
  long stride = (long)gridDim.x * blockDim.x;
  for (; i < n; i += stride) dst[i] = bf16_bits(src[i]);
}

__global__ void embed_kernel(const int* __restrict__ x,
                             const float* __restrict__ emb,
                             uint16_t* __restrict__ out, long n, int E) {
  long i = (long)blockIdx.x * blockDim.x + threadIdx.x;
  long stride = (long)gridDim.x * blockDim.x;
  for (; i < n; i += stride) {
    long bt = i / E;
    int  e  = (int)(i - bt * E);
    out[i] = bf16_bits(emb[(size_t)x[bt] * E + e]);
  }
}

__global__ void init_state_kernel(const float* __restrict__ hidden,
                                  const float* __restrict__ cell,
                                  float* h0f, float* h1f,
                                  uint16_t* h0b, uint16_t* h1b,
                                  float* c0, float* c1, int n) {
  int i = blockIdx.x * blockDim.x + threadIdx.x;
  if (i >= n) return;
  float h = hidden[i], c = cell[i];
  h0f[i] = h; h1f[i] = h;
  uint16_t hb = bf16_bits(h);
  h0b[i] = hb; h1b[i] = hb;
  c0[i] = c; c1[i] = c;
}

// -------------------------------------------------- generic bf16 WMMA GEMM ---
// C[m, n] = sum_k A[m,k] * W[n,k]  (+ bias[n]).
// One wave owns TWO adjacent 16x16 C tiles sharing the A fragment, and the K
// loop is software-pipelined: loads for chunk kk+1 are issued before the two
// independent WMMAs of chunk kk, so L2 latency overlaps matrix-op execution.
__global__ void gemm_bf16_kernel(const uint16_t* __restrict__ A,
                                 const uint16_t* __restrict__ W,
                                 const float* __restrict__ bias,
                                 float* __restrict__ C, long ldc,
                                 int N, int K) {
  const int wave = threadIdx.x >> 5;
  const int nt   = blockIdx.x * (blockDim.x >> 5) + wave;
  const int n0   = nt << 5;                  // two 16-wide column tiles
  if (n0 >= N) return;                       // wave-uniform: EXEC stays all-1s
  const int m0 = blockIdx.y << 4;
  const int KK = K >> 5;

  f32x8 z = {0.f, 0.f, 0.f, 0.f, 0.f, 0.f, 0.f, 0.f};
  f32x8 acc0 = z, acc1 = z;

  const uint16_t* Ab = A + (size_t)m0 * K;
  const uint16_t* W0 = W + (size_t)n0 * K;
  const uint16_t* W1 = W + (size_t)(n0 + 16) * K;
  // per-lane row pointers for divergent prefetch (covers all 16 tile rows)
  const int lane = threadIdx.x & 31;
  const uint16_t* pw0 = W0 + (size_t)(lane & 15) * K;
  const uint16_t* pw1 = W1 + (size_t)(lane & 15) * K;

  bf16x16 af = load_frag(Ab, K);
  bf16x16 b0 = load_frag(W0, K);
  bf16x16 b1 = load_frag(W1, K);
  for (int kk = 0; kk + 1 < KK; ++kk) {
    const size_t off = (size_t)(kk + 1) << 5;
    bf16x16 af_n = load_frag(Ab + off, K);
    bf16x16 b0_n = load_frag(W0 + off, K);
    bf16x16 b1_n = load_frag(W1 + off, K);
    if (kk + 4 < KK) {                       // pull K-chunk kk+4 toward WGP$
      __builtin_prefetch(pw0 + ((size_t)(kk + 4) << 5), 0, 1);
      __builtin_prefetch(pw1 + ((size_t)(kk + 4) << 5), 0, 1);
    }
    acc0 = __builtin_amdgcn_wmma_f32_16x16x32_bf16(false, af, false, b0,
                                                   (short)0, acc0, false, false);
    acc1 = __builtin_amdgcn_wmma_f32_16x16x32_bf16(false, af, false, b1,
                                                   (short)0, acc1, false, false);
    af = af_n; b0 = b0_n; b1 = b1_n;
  }
  acc0 = __builtin_amdgcn_wmma_f32_16x16x32_bf16(false, af, false, b0,
                                                 (short)0, acc0, false, false);
  acc1 = __builtin_amdgcn_wmma_f32_16x16x32_bf16(false, af, false, b1,
                                                 (short)0, acc1, false, false);

  const int col = lane & 15, hi = lane >> 4;
  const int na = n0 + col, nb = n0 + 16 + col;
  const float bva = bias ? bias[na] : 0.f;
  const float bvb = bias ? bias[nb] : 0.f;
#pragma unroll
  for (int r = 0; r < 8; ++r) {
    int m = m0 + r + (hi << 3);
    C[(size_t)m * ldc + na] = acc0[r] + bva;
    C[(size_t)m * ldc + nb] = acc1[r] + bvb;
  }
}

// -------------------------------- fused LSTM cell: 4-gate WMMA + epilogue ---
// gates = X @ Wih^T + Hp @ Whh^T + b_ih + b_hh ; i,f,g,o gate tiles share the
// same (m0,n0) so the sigmoid/tanh state update fuses into the C fragment.
// Four independent accumulators already give back-to-back WMMAs per K chunk.
__global__ void lstm_wmma_kernel(const uint16_t* __restrict__ X, int Kx,
                                 const uint16_t* __restrict__ Wih,
                                 const uint16_t* __restrict__ Hp, int Kh,
                                 const uint16_t* __restrict__ Whh,
                                 const float* __restrict__ b_ih,
                                 const float* __restrict__ b_hh,
                                 const float* __restrict__ c_prev,
                                 float* __restrict__ c_out,
                                 float* __restrict__ h_out_f,
                                 uint16_t* __restrict__ h_out_bf,
                                 uint16_t* __restrict__ h_bf2, long ld2) {
  constexpr int H = 512;
  const int n0 = blockIdx.x << 4;            // column tile inside each gate
  const int m0 = blockIdx.y << 4;            // batch tile

  f32x8 z = {0.f, 0.f, 0.f, 0.f, 0.f, 0.f, 0.f, 0.f};
  f32x8 acc[4] = {z, z, z, z};

  const uint16_t* Xb = X + (size_t)m0 * Kx;
  for (int kk = 0; kk < (Kx >> 5); ++kk) {
    bf16x16 af = load_frag(Xb + ((size_t)kk << 5), Kx);
#pragma unroll
    for (int g = 0; g < 4; ++g) {
      bf16x16 bf = load_frag(Wih + (size_t)(g * H + n0) * Kx + ((size_t)kk << 5), Kx);
      acc[g] = __builtin_amdgcn_wmma_f32_16x16x32_bf16(false, af, false, bf,
                                                       (short)0, acc[g], false, false);
    }
  }
  const uint16_t* Hb = Hp + (size_t)m0 * Kh;
  for (int kk = 0; kk < (Kh >> 5); ++kk) {
    bf16x16 af = load_frag(Hb + ((size_t)kk << 5), Kh);
#pragma unroll
    for (int g = 0; g < 4; ++g) {
      bf16x16 bf = load_frag(Whh + (size_t)(g * H + n0) * Kh + ((size_t)kk << 5), Kh);
      acc[g] = __builtin_amdgcn_wmma_f32_16x16x32_bf16(false, af, false, bf,
                                                       (short)0, acc[g], false, false);
    }
  }

  const int lane = threadIdx.x & 31;
  const int col  = lane & 15, hi = lane >> 4;
  const int hc   = n0 + col;
  const float bi0 = b_ih[hc]         + b_hh[hc];
  const float bi1 = b_ih[H + hc]     + b_hh[H + hc];
  const float bi2 = b_ih[2 * H + hc] + b_hh[2 * H + hc];
  const float bi3 = b_ih[3 * H + hc] + b_hh[3 * H + hc];
#pragma unroll
  for (int r = 0; r < 8; ++r) {
    const int m = m0 + r + (hi << 3);
    const size_t idx = (size_t)m * H + hc;
    float iv = sigmoidf_(acc[0][r] + bi0);
    float fv = sigmoidf_(acc[1][r] + bi1);
    float gv = tanhf(acc[2][r] + bi2);
    float ov = sigmoidf_(acc[3][r] + bi3);
    float cn = fv * c_prev[idx] + iv * gv;
    float hn = ov * tanhf(cn);
    c_out[idx]   = cn;
    h_out_f[idx] = hn;
    uint16_t hb = bf16_bits(hn);
    h_out_bf[idx] = hb;
    if (h_bf2) h_bf2[(size_t)m * ld2 + hc] = hb;
  }
}

// ------------------------------------- attention (per batch row workgroup) ---
__global__ void attention_kernel(const float* __restrict__ h1,     // [B,H]
                                 const float* __restrict__ Wdec,   // [H,H]
                                 const float* __restrict__ encp,   // [B,S,H]
                                 const float* __restrict__ enco,   // [B,S,H]
                                 const float* __restrict__ vatt,   // [H]
                                 const int*   __restrict__ mask,   // [B,S]
                                 const uint16_t* __restrict__ embedded, // [B,T,E]
                                 int t,
                                 float* __restrict__ attn_out,     // [B,T,S]
                                 uint16_t* __restrict__ lstm_in,   // [B,E+H]
                                 uint16_t* __restrict__ h1ctx) {   // [B,2H]
  constexpr int H = 512, S = 128, E = 256, T = 64;
  const int b   = blockIdx.x;
  const int tid = threadIdx.x;               // 256 threads = 8 waves

  __shared__ float s_dp[H];
  __shared__ float s_sc[S];

  // dec_proj = h1[b] @ Wdec^T
  const float* hr = h1 + (size_t)b * H;
  for (int j = tid; j < H; j += 256) {
    const float* wr = Wdec + (size_t)j * H;
    float a = 0.f;
    for (int k = 0; k < H; ++k) a += hr[k] * wr[k];
    s_dp[j] = a;
  }
  __syncthreads();

  // scores[s] = v_att . tanh(enc_proj[b,s,:] + dec_proj)
  const int wave = tid >> 5, lane = tid & 31;
  for (int s = wave; s < S; s += 8) {
    const float* er = encp + ((size_t)b * S + s) * H;
    float p = 0.f;
    for (int h = lane; h < H; h += 32) p += tanhf(er[h] + s_dp[h]) * vatt[h];
#pragma unroll
    for (int off = 16; off > 0; off >>= 1) p += __shfl_xor(p, off, 32);
    if (lane == 0)
      s_sc[s] = (mask[(size_t)b * S + s] != 0) ? p : -1e9f;
  }
  __syncthreads();

  // softmax over S (every thread redundantly reduces -> deterministic)
  float mx = -3.4e38f;
  for (int s = 0; s < S; ++s) mx = fmaxf(mx, s_sc[s]);
  __syncthreads();
  if (tid < S) s_sc[tid] = __expf(s_sc[tid] - mx);
  __syncthreads();
  float sum = 0.f;
  for (int s = 0; s < S; ++s) sum += s_sc[s];
  __syncthreads();
  if (tid < S) {
    float a = s_sc[tid] / sum;
    s_sc[tid] = a;
    attn_out[((size_t)b * T + t) * S + tid] = a;
  }
  __syncthreads();

  // context = attn . encoder_outputs ; write bf16 into GEMM input buffers
  for (int h = tid; h < H; h += 256) {
    const float* eb = enco + (size_t)b * S * H + h;
    float c = 0.f;
    for (int s = 0; s < S; ++s) c += s_sc[s] * eb[(size_t)s * H];
    uint16_t cb = bf16_bits(c);
    lstm_in[(size_t)b * (E + H) + E + h] = cb;
    h1ctx [(size_t)b * (2 * H) + H + h]  = cb;
  }
  // embedded token slice (E == blockDim.x)
  lstm_in[(size_t)b * (E + H) + tid] = embedded[((size_t)b * T + t) * E + tid];
}

// ------------------------------------------------------------------- host ---
extern "C" void kernel_launch(void* const* d_in, const int* in_sizes, int n_in,
                              void* d_out, int out_size, void* d_ws, size_t ws_size,
                              hipStream_t stream) {
  (void)in_sizes; (void)n_in; (void)out_size; (void)ws_size;
  constexpr int B = 32, T = 64, S = 128, H = 512, E = 256, V = 16000;

  const int*   x     = (const int*)  d_in[0];
  const float* hidden= (const float*)d_in[1];
  const float* cell  = (const float*)d_in[2];
  const float* enco  = (const float*)d_in[3];
  const int*   emask = (const int*)  d_in[4];
  const float* emb   = (const float*)d_in[5];
  const float* W_enc = (const float*)d_in[6];
  const float* W_dec = (const float*)d_in[7];
  const float* v_att = (const float*)d_in[8];
  const float* W_ih0 = (const float*)d_in[9];
  const float* W_hh0 = (const float*)d_in[10];
  const float* b_ih0 = (const float*)d_in[11];
  const float* b_hh0 = (const float*)d_in[12];
  const float* W_ih1 = (const float*)d_in[13];
  const float* W_hh1 = (const float*)d_in[14];
  const float* b_ih1 = (const float*)d_in[15];
  const float* b_hh1 = (const float*)d_in[16];
  const float* fc_W  = (const float*)d_in[17];
  const float* fc_b  = (const float*)d_in[18];

  float* pred      = (float*)d_out;                       // [B,T,V]
  float* attn_base = pred + (size_t)B * T * V;            // [B,T,S]

  // workspace bump allocator (256B aligned)
  char* wp = (char*)d_ws;
  auto alloc = [&](size_t bytes) -> char* {
    char* p = wp; wp += (bytes + 255) & ~(size_t)255; return p;
  };
  uint16_t* fcWb  = (uint16_t*)alloc((size_t)V * 2 * H * 2);
  uint16_t* Wih0b = (uint16_t*)alloc((size_t)4 * H * (E + H) * 2);
  uint16_t* Whh0b = (uint16_t*)alloc((size_t)4 * H * H * 2);
  uint16_t* Wih1b = (uint16_t*)alloc((size_t)4 * H * H * 2);
  uint16_t* Whh1b = (uint16_t*)alloc((size_t)4 * H * H * 2);
  uint16_t* Wencb = (uint16_t*)alloc((size_t)H * H * 2);
  uint16_t* encb  = (uint16_t*)alloc((size_t)B * S * H * 2);
  uint16_t* embb  = (uint16_t*)alloc((size_t)B * T * E * 2);
  float*    encp  = (float*)   alloc((size_t)B * S * H * 4);
  float*    h0f[2] = {(float*)alloc((size_t)B * H * 4), (float*)alloc((size_t)B * H * 4)};
  float*    h1f[2] = {(float*)alloc((size_t)B * H * 4), (float*)alloc((size_t)B * H * 4)};
  uint16_t* h0b[2] = {(uint16_t*)alloc((size_t)B * H * 2), (uint16_t*)alloc((size_t)B * H * 2)};
  uint16_t* h1b[2] = {(uint16_t*)alloc((size_t)B * H * 2), (uint16_t*)alloc((size_t)B * H * 2)};
  float*    c0 = (float*)alloc((size_t)B * H * 4);
  float*    c1 = (float*)alloc((size_t)B * H * 4);
  uint16_t* lstm_in = (uint16_t*)alloc((size_t)B * (E + H) * 2);
  uint16_t* h1ctx   = (uint16_t*)alloc((size_t)B * 2 * H * 2);

  auto cvt = [&](const float* s, uint16_t* d, long n) {
    int blocks = (int)((n + 1023) / 1024);
    cvt_bf16_kernel<<<blocks, 256, 0, stream>>>(s, d, n);
  };
  // one-time (per launch) weight/activation down-conversions to bf16
  cvt(fc_W,  fcWb,  (long)V * 2 * H);
  cvt(W_ih0, Wih0b, (long)4 * H * (E + H));
  cvt(W_hh0, Whh0b, (long)4 * H * H);
  cvt(W_ih1, Wih1b, (long)4 * H * H);
  cvt(W_hh1, Whh1b, (long)4 * H * H);
  cvt(W_enc, Wencb, (long)H * H);
  cvt(enco,  encb,  (long)B * S * H);

  {
    long n = (long)B * T * E;
    embed_kernel<<<(int)((n + 1023) / 1024), 256, 0, stream>>>(x, emb, embb, n, E);
  }
  init_state_kernel<<<(B * H + 255) / 256, 256, 0, stream>>>(
      hidden, cell, h0f[0], h1f[0], h0b[0], h1b[0], c0, c1, B * H);

  // step-invariant: enc_proj = encoder_outputs @ W_enc^T   (M=B*S, N=H, K=H)
  // N tiles-of-32 = 16 ; 4 waves/block -> grid.x = 4
  gemm_bf16_kernel<<<dim3(4, (B * S) / 16), 128, 0, stream>>>(
      encb, Wencb, nullptr, encp, (long)H, H, H);

  for (int t = 0; t < T; ++t) {
    const int cur = t & 1, nxt = cur ^ 1;

    attention_kernel<<<B, 256, 0, stream>>>(
        h1f[cur], W_dec, encp, enco, v_att, emask, embb, t,
        attn_base, lstm_in, h1ctx);

    lstm_wmma_kernel<<<dim3(H / 16, B / 16), 32, 0, stream>>>(
        lstm_in, E + H, Wih0b, h0b[cur], H, Whh0b, b_ih0, b_hh0,
        c0, c0, h0f[nxt], h0b[nxt], nullptr, 0);

    lstm_wmma_kernel<<<dim3(H / 16, B / 16), 32, 0, stream>>>(
        h0b[nxt], H, Wih1b, h1b[cur], H, Whh1b, b_ih1, b_hh1,
        c1, c1, h1f[nxt], h1b[nxt], h1ctx, (long)2 * H);

    // fc: N tiles-of-32 = 500 ; 4 waves/block -> grid.x = 125, grid.y = 2
    gemm_bf16_kernel<<<dim3(125, B / 16), 128, 0, stream>>>(
        h1ctx, fcWb, fc_b, pred + (size_t)t * V, (long)T * V, V, 2 * H);
  }
}
